// LaneAtt_67413806678154
// MI455X (gfx1250) — matrix operand order repaired and checked by reading, not compiled, not run on hardware
//
#include <hip/hip_runtime.h>
#include <hip/hip_bf16.h>

// ---------------- problem constants ----------------
#define PB    2784          // number of anchors (P)
#define DD    704           // AFC * FH = 64*11
#define BN    8             // batch
#define FHH   11
#define FWW   20
#define BBC   512
#define AFCC  64
#define NPOS  (FHH*FWW)     // 220
#define NREG  73            // NUM_OFFSETS+1
#define NOUTW 75            // 73 reg rows + 2 cls rows
#define LDCOUT 80           // padded ldc for reg/cls output
#define ATTW_LDB 2816       // att_w^T padded column count (mult of 64)
#define RWT_LDB  128        // [reg;cls]^T padded column count

#define APITCH 20           // dwords per A-tile row   (16 data + pad, 80B: 16B aligned)
#define BPITCH 36           // dwords per B [k][n] row (32 data + pad, 144B: 16B aligned)

typedef __attribute__((ext_vector_type(16))) __bf16 bf16x16;
typedef __attribute__((ext_vector_type(8)))  float  floatx8;
typedef __attribute__((ext_vector_type(4)))  int    v4i;

union AFrag { bf16x16 v; unsigned u[8]; };

#define HAS_ASYNC_LDS __has_builtin(__builtin_amdgcn_global_load_async_to_lds_b128)

#if HAS_ASYNC_LDS
__device__ __forceinline__ void g2l_async16(const void* g, void* l) {
  v4i* gt = (v4i*)const_cast<void*>(g);
  v4i* lt = (v4i*)l;
  __builtin_amdgcn_global_load_async_to_lds_b128(
      (__attribute__((address_space(1))) v4i*)gt,
      (__attribute__((address_space(3))) v4i*)lt, 0, 0);
}
#endif

template <int N>
__device__ __forceinline__ void wait_async() {
#if __has_builtin(__builtin_amdgcn_s_wait_asynccnt)
  __builtin_amdgcn_s_wait_asynccnt(N);
#else
  if constexpr (N == 0) asm volatile("s_wait_asynccnt 0x0" ::: "memory");
  else                  asm volatile("s_wait_asynccnt 0x3" ::: "memory");
#endif
}

__device__ __forceinline__ unsigned short f2bf(float f) {
  unsigned u = __float_as_uint(f);
  unsigned r = u + 0x7fffu + ((u >> 16) & 1u);   // round-to-nearest-even
  return (unsigned short)(r >> 16);
}

// ---------------- weight conversion (transpose to [K][N], bf16, padded) ----------------
__global__ __launch_bounds__(256) void build_attwT(const float* __restrict__ att_w,
                                                   unsigned short* __restrict__ wT) {
  int i = blockIdx.x * 256 + threadIdx.x;        // over DD * ATTW_LDB
  if (i >= DD * ATTW_LDB) return;
  int d = i / ATTW_LDB, n = i % ATTW_LDB;
  float v = (n < PB - 1) ? att_w[(size_t)n * DD + d] : 0.f;
  wT[i] = f2bf(v);
}

__global__ __launch_bounds__(256) void build_rwbT(const float* __restrict__ reg_w,
                                                  const float* __restrict__ cls_w,
                                                  unsigned short* __restrict__ wT) {
  int i = blockIdx.x * 256 + threadIdx.x;        // over 2*DD * RWT_LDB
  if (i >= 2 * DD * RWT_LDB) return;
  int k = i / RWT_LDB, n = i % RWT_LDB;
  float v = 0.f;
  if (n < NREG)       v = reg_w[(size_t)n * 2 * DD + k];
  else if (n < NOUTW) v = cls_w[(size_t)(n - NREG) * 2 * DD + k];
  wT[i] = f2bf(v);
}

// ---------------- 1x1 conv ----------------
__global__ __launch_bounds__(64) void conv1x1(const float* __restrict__ x,
                                              const float* __restrict__ w,
                                              const float* __restrict__ bias,
                                              float* __restrict__ f) {
  int flat = blockIdx.x;                 // 0 .. B*NPOS-1
  int b = flat / NPOS, pos = flat % NPOS;
  __shared__ float xs[BBC];
  const float* xp = x + (size_t)b * BBC * NPOS + pos;
  for (int c = threadIdx.x; c < BBC; c += 64) xs[c] = xp[(size_t)c * NPOS];
  __syncthreads();
  int o = threadIdx.x;
  const float* wr = w + (size_t)o * BBC;
  float acc = 0.f;
#pragma unroll 8
  for (int c = 0; c < BBC; ++c) acc += xs[c] * wr[c];
  f[((size_t)b * AFCC + o) * NPOS + pos] = acc + bias[o];
}

// ---------------- gather + mask -> feat (bf16) ----------------
__global__ __launch_bounds__(128) void gather_feat(const float* __restrict__ f,
                                                   const int* __restrict__ cut_x,
                                                   const unsigned char* __restrict__ invalid,
                                                   unsigned short* __restrict__ featb) {
  int bp = blockIdx.x;                  // 0 .. B*P-1
  int b = bp / PB, p = bp % PB;
  __shared__ int cx[FHH];
  __shared__ unsigned char inv[FHH];
  if (threadIdx.x < FHH) {
    cx[threadIdx.x]  = cut_x[p * FHH + threadIdx.x];
    inv[threadIdx.x] = invalid[p * FHH + threadIdx.x];
  }
  __syncthreads();
  for (int d = threadIdx.x; d < DD; d += 128) {
    int a = d / FHH, h = d % FHH;
    float v = inv[h] ? 0.f : f[(((size_t)b * AFCC + a) * FHH + h) * FWW + cx[h]];
    featb[(size_t)bp * DD + d] = f2bf(v);
  }
}

// ---------------- bf16 WMMA GEMM, 64x64 block tile, K-tile 32 ----------------
// C[m,n] = sum_k A[m,k]*B[k,n] (+bias[n]).   B row-major [K][N], stride ldb (mult of 8).
//   SPLIT: A element k comes from A0 if k<ksplit else A1 (same stride lda).
//   K, ksplit multiples of 32.  Staged loads clamp; stores are guarded.
template <bool SPLIT>
__global__ __launch_bounds__(128) void gemm_bf16_wmma(
    const unsigned short* __restrict__ A0, const unsigned short* __restrict__ A1,
    int lda, int ksplit,
    const unsigned short* __restrict__ Bm, int ldb,
    const float* __restrict__ bias,
    float* __restrict__ Cf, unsigned short* __restrict__ Cb, int ldc,
    int M, int N, int K) {
  __shared__ __align__(16) unsigned a_tile[2][64 * APITCH];   // [m][ku]
  __shared__ __align__(16) unsigned b_tile[2][32 * BPITCH];   // [k][nu]
  const int tid  = threadIdx.x;
  const int wave = tid >> 5;
  const int lane = tid & 31;
  const int mblk = blockIdx.y * 64;
  const int nblk = blockIdx.x * 64;
  const int kspu = ksplit >> 1;
  const int nkt  = K >> 5;

  // 16B staging chunks
  auto a_src = [&](int kt, int c) -> const unsigned short* {   // c in 0..255
    int row = c >> 2, q = c & 3;
    int m = mblk + row; if (m >= M) m = M - 1;
    int kuo = kt * 16 + q * 4;          // dword offset within row
    if (SPLIT && kuo >= kspu) return A1 + (size_t)m * lda + (size_t)(kuo - kspu) * 2;
    return A0 + (size_t)m * lda + (size_t)kuo * 2;
  };
  auto b_src = [&](int kt, int c) -> const unsigned short* {   // c in 0..127
    int k = c >> 2, q = c & 3;
    return Bm + (size_t)(kt * 32 + k) * ldb + (size_t)(nblk + q * 8);
  };
  auto a_dst = [&](int buf, int c) -> unsigned* {
    int row = c >> 2, q = c & 3;
    return &a_tile[buf][row * APITCH + q * 4];
  };
  auto b_dst = [&](int buf, int c) -> unsigned* {
    int k = c >> 2, q = c & 3;
    return &b_tile[buf][k * BPITCH + q * 4];
  };

#if HAS_ASYNC_LDS
  auto stage = [&](int kt, int buf) {   // 3 async wave-ops: 512B each
#pragma unroll
    for (int i = 0; i < 2; ++i) g2l_async16(a_src(kt, tid + i * 128), a_dst(buf, tid + i * 128));
    g2l_async16(b_src(kt, tid), b_dst(buf, tid));
  };
#else
  auto stage = [&](int kt, int buf) {   // fallback: reg-staged uint4 copies
    uint4 av[2];
#pragma unroll
    for (int i = 0; i < 2; ++i) av[i] = *(const uint4*)a_src(kt, tid + i * 128);
#pragma unroll
    for (int i = 0; i < 2; ++i) *(uint4*)a_dst(buf, tid + i * 128) = av[i];
    *(uint4*)b_dst(buf, tid) = *(const uint4*)b_src(kt, tid);
  };
#endif

  floatx8 acc[4];
#pragma unroll
  for (int nt = 0; nt < 4; ++nt)
#pragma unroll
    for (int r = 0; r < 8; ++r) acc[nt][r] = 0.f;

  auto compute = [&](int buf) {
    // A fragment (16x32 bf16): lane L -> row M=L&15, K half sel by L>>4 (ISA 7.12.2)
    AFrag af;
    int arow = wave * 16 + (lane & 15);
    int koff = (lane >> 4) << 2;
#pragma unroll
    for (int t = 0; t < 8; ++t) {
      int idx = (t & 3) + ((t >> 2) << 3) + koff;
      af.u[t] = a_tile[buf][arow * APITCH + idx];
    }
    // B fragments (32x16 bf16): lane L -> column N=L&15, K half sel by L>>4 (same k order as A)
#pragma unroll
    for (int nt = 0; nt < 4; ++nt) {
      AFrag bf;
#pragma unroll
      for (int t = 0; t < 16; ++t) ((unsigned short*)bf.u)[t] = ((const unsigned short*)b_tile[buf])[((t & 7) + ((t >> 3) << 4) + ((lane >> 4) << 3)) * (BPITCH * 2) + nt * 16 + (lane & 15)];
      acc[nt] = __builtin_amdgcn_wmma_f32_16x16x32_bf16(
          false, af.v, false, bf.v, (short)0, acc[nt], false, false);
    }
  };

#if HAS_ASYNC_LDS
  stage(0, 0);                          // prologue
  for (int kt = 0; kt < nkt; ++kt) {
    if (kt + 1 < nkt) { stage(kt + 1, (kt + 1) & 1); wait_async<3>(); }
    else              { wait_async<0>(); }
    __syncthreads();                    // tile kt resident for all waves
    compute(kt & 1);
    __syncthreads();                    // reads done; buffer reusable
  }
#else
  for (int kt = 0; kt < nkt; ++kt) {
    stage(kt, 0);
    __syncthreads();
    compute(0);
    __syncthreads();
  }
#endif

  // epilogue: C/D f32 layout: lane = N | ((m>=8)<<4), vgpr r = m&7
  int nloc  = lane & 15;
  int mbase = (lane >> 4) * 8;
#pragma unroll
  for (int nt = 0; nt < 4; ++nt) {
#pragma unroll
    for (int r = 0; r < 8; ++r) {
      int m = mblk + wave * 16 + mbase + r;
      int n = nblk + nt * 16 + nloc;
      if (m < M && n < N) {
        float v = acc[nt][r];
        if (bias) v += bias[n];
        if (Cf) Cf[(size_t)m * ldc + n] = v;
        if (Cb) Cb[(size_t)m * ldc + n] = f2bf(v);
      }
    }
  }
}

// ---------------- softmax over P-1 scores + diagonal-zero insertion ----------------
__global__ __launch_bounds__(256) void softmax_diag(const float* __restrict__ scores,
                                                    unsigned short* __restrict__ attb) {
  int i = blockIdx.x;                    // anchor row 0..P-1
  __shared__ float red[256];
  const float* s = scores + (size_t)i * (PB - 1);
  float m = -1e30f;
  for (int k = threadIdx.x; k < PB - 1; k += 256) m = fmaxf(m, s[k]);
  red[threadIdx.x] = m; __syncthreads();
  for (int st = 128; st > 0; st >>= 1) {
    if (threadIdx.x < st) red[threadIdx.x] = fmaxf(red[threadIdx.x], red[threadIdx.x + st]);
    __syncthreads();
  }
  m = red[0]; __syncthreads();
  float sum = 0.f;
  for (int k = threadIdx.x; k < PB - 1; k += 256) sum += __expf(s[k] - m);
  red[threadIdx.x] = sum; __syncthreads();
  for (int st = 128; st > 0; st >>= 1) {
    if (threadIdx.x < st) red[threadIdx.x] += red[threadIdx.x + st];
    __syncthreads();
  }
  float inv = 1.f / red[0];
  unsigned short* arow = attb + (size_t)i * PB;
  for (int j = threadIdx.x; j < PB; j += 256) {
    float v;
    if (j == i) v = 0.f;
    else { int k = j - (j > i); v = __expf(s[k] - m) * inv; }
    arow[j] = f2bf(v);
  }
}

// ---------------- assemble outputs ----------------
__global__ __launch_bounds__(96) void assemble(const float* __restrict__ regout,
                                               const float* __restrict__ anchors,
                                               const float* __restrict__ reg_b,
                                               const float* __restrict__ cls_b,
                                               float* __restrict__ out) {
  int bp = blockIdx.x;                   // 0..B*P-1
  int p  = bp % PB;
  const float* rr = regout + (size_t)bp * LDCOUT;
  const float* an = anchors + (size_t)p * (2 + 72);
  float* out0 = out;                                      // (B,P,72)
  float* out1 = out + (size_t)BN * PB * 72;               // (B,P)
  float* out2 = out1 + (size_t)BN * PB;                   // (B,P)
  float* out3 = out2 + (size_t)BN * PB;                   // (B,P,2)
  int j = threadIdx.x;
  if (j < 72)       out0[(size_t)bp * 72 + j] = an[2 + j] + rr[1 + j] + reg_b[1 + j];
  else if (j == 72) out1[bp] = an[0];
  else if (j == 73) out2[bp] = an[1] + rr[0] + reg_b[0];
  else if (j < 76)  out3[(size_t)bp * 2 + (j - 74)] = rr[NREG + (j - 74)] + cls_b[j - 74];
}

// ---------------- host orchestration ----------------
extern "C" void kernel_launch(void* const* d_in, const int* in_sizes, int n_in,
                              void* d_out, int out_size, void* d_ws, size_t ws_size,
                              hipStream_t stream) {
  (void)in_sizes; (void)n_in; (void)out_size; (void)ws_size;
  const float* x      = (const float*)d_in[0];
  const float* conv_w = (const float*)d_in[1];
  const float* conv_b = (const float*)d_in[2];
  const float* att_w  = (const float*)d_in[3];
  const float* att_b  = (const float*)d_in[4];
  const float* cls_w  = (const float*)d_in[5];
  const float* cls_b  = (const float*)d_in[6];
  const float* reg_w  = (const float*)d_in[7];
  const float* reg_b  = (const float*)d_in[8];
  const float* anchors = (const float*)d_in[9];
  const int*   cut_x  = (const int*)d_in[10];
  const unsigned char* invalid = (const unsigned char*)d_in[11];
  float* out = (float*)d_out;

  // workspace carve-up (256B aligned)
  char* ws = (char*)d_ws;
  size_t off = 0;
  auto carve = [&](size_t bytes) -> char* {
    char* p = ws + off;
    off = (off + bytes + 255) & ~(size_t)255;
    return p;
  };
  float*          f_buf    = (float*)         carve((size_t)BN * AFCC * NPOS * 4);
  unsigned short* featb    = (unsigned short*)carve((size_t)BN * PB * DD * 2);
  unsigned short* att_wT   = (unsigned short*)carve((size_t)DD * ATTW_LDB * 2);
  unsigned short* rwbT     = (unsigned short*)carve((size_t)2 * DD * RWT_LDB * 2);
  float*          scores_b = (float*)         carve((size_t)PB * (PB - 1) * 4);
  unsigned short* attb_b   = (unsigned short*)carve((size_t)PB * PB * 2);
  unsigned short* attfeatb = (unsigned short*)carve((size_t)BN * PB * DD * 2);
  float*          regout   = (float*)         carve((size_t)BN * PB * LDCOUT * 4);

  // one-shot weight transposition + bf16 conversion
  {
    int n = DD * ATTW_LDB;
    build_attwT<<<(n + 255) / 256, 256, 0, stream>>>(att_w, att_wT);
    int m = 2 * DD * RWT_LDB;
    build_rwbT<<<(m + 255) / 256, 256, 0, stream>>>(reg_w, cls_w, rwbT);
  }
  // conv + gather
  conv1x1<<<BN * NPOS, 64, 0, stream>>>(x, conv_w, conv_b, f_buf);
  gather_feat<<<BN * PB, 128, 0, stream>>>(f_buf, cut_x, invalid, featb);

  // per-batch attention pipeline (reuses scores/att buffers)
  const int mtiles  = (PB + 63) / 64;         // 44
  const int ntilesS = (PB - 1 + 63) / 64;     // 44
  for (int b = 0; b < BN; ++b) {
    const unsigned short* fb = featb + (size_t)b * PB * DD;
    // scores = feat @ att_w^T + att_b    (M=P, N=P-1, K=D)
    gemm_bf16_wmma<false><<<dim3(ntilesS, mtiles), 128, 0, stream>>>(
        fb, fb, DD, DD, att_wT, ATTW_LDB, att_b,
        scores_b, (unsigned short*)nullptr, PB - 1, PB, PB - 1, DD);
    // softmax + diagonal insertion -> att (bf16, P x P)
    softmax_diag<<<PB, 256, 0, stream>>>(scores_b, attb_b);
    // att_feat = att @ feat              (M=P, N=D, K=P)
    gemm_bf16_wmma<false><<<dim3(DD / 64, mtiles), 128, 0, stream>>>(
        attb_b, attb_b, PB, PB, fb, DD, (const float*)nullptr,
        (float*)nullptr, attfeatb + (size_t)b * PB * DD, DD, PB, DD, PB);
  }

  // reg+cls = [att_feat | feat] @ [reg_w;cls_w]^T   (M=B*P, N=75, K=2D, split-K A)
  gemm_bf16_wmma<true><<<dim3((NOUTW + 63) / 64, (BN * PB) / 64), 128, 0, stream>>>(
      attfeatb, featb, DD, DD, rwbT, RWT_LDB, (const float*)nullptr,
      regout, (unsigned short*)nullptr, LDCOUT, BN * PB, NOUTW, 2 * DD);

  // final outputs
  assemble<<<BN * PB, 96, 0, stream>>>(regout, anchors, reg_b, cls_b, out);
}
